// LTCNLayer_26594437497394
// MI455X (gfx1250) — matrix operand order repaired and checked by reading, not compiled, not run on hardware
//
#include <hip/hip_runtime.h>
#include <stdint.h>

// ---------------------------------------------------------------------------
// LTC layer for MI455X (gfx1250, wave32, WMMA + async global->LDS).
//   BATCH=4096, N=4096 (NB=64 blocks of K=64), IN_DIM=256
// Memory-bound (AI ~64 FLOP/B << f16-WMMA balance): stream y once, hold
// per-block weights in LDS as f16 (pre-converted once into d_ws, DMA'd with
// global_load_async_to_lds_b128), do all GEMMs with v_wmma_f32_16x16x32_f16.
// ---------------------------------------------------------------------------

typedef __attribute__((ext_vector_type(16))) _Float16 v16h;
typedef __attribute__((ext_vector_type(8)))  _Float16 v8h;
typedef __attribute__((ext_vector_type(4)))  _Float16 v4h;
typedef __attribute__((ext_vector_type(8)))  float    v8f;
typedef __attribute__((ext_vector_type(4)))  float    v4f;

#define IN_DIM_  256
#define KDIM     64
#define NBLK     64
#define NTOT     4096
#define BATCH_   4096
#define DT_      0.05f
#define TAU_EPS_ 1e-6f
#define BT       128     // batch rows per workgroup
#define NTHREADS 256     // 8 wave32

// f16-element offsets inside the workspace
#define WS_WIN   0u                               // 64*256
#define WS_WFWD  (WS_WIN  + 64u*256u)             // 63*64*64
#define WS_WREC  (WS_WFWD + 63u*64u*64u)          // 64*64*64
#define WS_EL    (WS_WREC + 64u*64u*64u)
#define WS_ELR   (WS_EL   + 64u*64u*64u)
#define WS_UT    (WS_ELR  + 64u*64u*64u)          // 4096*256
#define WS_ELEMS ((size_t)WS_UT + (size_t)BATCH_ * IN_DIM_)
#define WS_BYTES (WS_ELEMS * 2)

// ---- CDNA5 async global->LDS copy (bypasses VGPRs, tracked by ASYNCcnt) ----
__device__ __forceinline__ void async_b128(unsigned lds_off, unsigned long long gbase,
                                           unsigned goff) {
    asm volatile("global_load_async_to_lds_b128 %0, %1, %2 offset:0"
                 :: "v"(lds_off), "v"(goff), "s"(gbase) : "memory");
}
__device__ __forceinline__ void async_wait0() {
    asm volatile("s_wait_asynccnt 0x0" ::: "memory");
}
__device__ __forceinline__ unsigned lds_off_of(const void* p) {
    return (unsigned)(uintptr_t)p;   // LDS aperture: addr[31:0] = LDS byte offset
}

// ---- WMMA fragment loaders (f16, LDS row-major source) --------------------
// A (16x32, MxK): lane 0-15 -> row M=lane, K=0..15 ; lane 16-31 -> K=16..31
__device__ __forceinline__ v16h load_a_frag(const _Float16* A, int lda,
                                            int m0, int k0, int lane) {
    const int m  = m0 + (lane & 15);
    const int kb = (lane >> 4) * 16;
    const _Float16* p = A + m * lda + k0 + kb;
    v8h lo = *(const v8h*)(p);
    v8h hi = *(const v8h*)(p + 8);
    v16h a;
#pragma unroll
    for (int i = 0; i < 8; ++i) { a[i] = lo[i]; a[8 + i] = hi[i]; }
    return a;
}
// B (32x16, KxN) = W^T with W row-major [N][K]: lane reads 16 contiguous f16.
__device__ __forceinline__ v16h load_b_frag(const _Float16* W, int ldw,
                                            int n0, int k0, int lane) {
    const int n  = n0 + (lane & 15);
    const int kb = (lane >> 4) * 16;
    return *(const v16h*)(W + n * ldw + k0 + kb);
}
__device__ __forceinline__ v8f wmma_acc(const _Float16* A, int lda,
                                        const _Float16* W, int ldw,
                                        int m0, int n0, int Kd, int lane, v8f acc) {
#pragma unroll
    for (int k0 = 0; k0 < Kd; k0 += 32) {
        v16h a = load_a_frag(A, lda, m0, k0, lane);
        v16h b = load_b_frag(W, ldw, n0, k0, lane);
        acc = __builtin_amdgcn_wmma_f32_16x16x32_f16(false, a, false, b,
                                                     (short)0, acc, false, false);
    }
    return acc;
}

__device__ __forceinline__ float softplusf(float x) {
    return fmaxf(x, 0.0f) + log1pf(__expf(-fabsf(x)));
}

// ---- pre-pass: one-time f32 -> f16 conversion into workspace ---------------
__global__ void cvt_kernel(const float* __restrict__ src, _Float16* __restrict__ dst,
                           int n4) {
    int i = blockIdx.x * blockDim.x + threadIdx.x;
    if (i < n4) {
        v4f s = ((const v4f*)src)[i];
        v4h d;
#pragma unroll
        for (int k = 0; k < 4; ++k) d[k] = (_Float16)s[k];
        ((v4h*)dst)[i] = d;
    }
}

// ---- main kernel -----------------------------------------------------------
__global__ __launch_bounds__(NTHREADS, 1)
void ltcn_kernel(const float* __restrict__ y,      const float* __restrict__ u_t,
                 const float* __restrict__ tau_raw,
                 const float* __restrict__ W_in_w, const float* __restrict__ W_in_b,
                 const float* __restrict__ W_fwd_w,const float* __restrict__ W_fwd_b,
                 const float* __restrict__ W_rec_w,const float* __restrict__ W_rec_b,
                 const float* __restrict__ E_l,    const float* __restrict__ E_l_r,
                 const _Float16* __restrict__ wsh, int use_ws,
                 float* __restrict__ out)
{
    __shared__ __attribute__((aligned(16))) _Float16 sW0 [KDIM * KDIM]; // Wfwd / Win chunk
    __shared__ __attribute__((aligned(16))) _Float16 sA0 [BT   * KDIM]; // prev-y / u_t chunk
    __shared__ __attribute__((aligned(16))) _Float16 sWr [KDIM * KDIM];
    __shared__ __attribute__((aligned(16))) _Float16 sEl [KDIM * KDIM];
    __shared__ __attribute__((aligned(16))) _Float16 sElr[KDIM * KDIM];
    __shared__ __attribute__((aligned(16))) _Float16 sY  [BT   * KDIM];
    __shared__ __attribute__((aligned(16))) _Float16 sNo [BT   * KDIM];
    __shared__ __attribute__((aligned(16))) _Float16 sNr [BT   * KDIM];

    const int j    = blockIdx.x;         // neuron block 0..63
    const int b0   = blockIdx.y * BT;    // batch-tile origin
    const int tid  = threadIdx.x;
    const int lane = tid & 31;
    const int wave = tid >> 5;           // 0..7 -> m-tile
    const int m0   = wave * 16;
    const int nl   = lane & 15;
    const int mh   = (lane >> 4) * 8;

    // ---------------- stage weights (async f16 DMA or sync cvt) ------------
    if (use_ws) {
        const unsigned long long bWr = (unsigned long long)(uintptr_t)(wsh + WS_WREC + (size_t)j * 4096);
        const unsigned long long bEl = (unsigned long long)(uintptr_t)(wsh + WS_EL   + (size_t)j * 4096);
        const unsigned long long bEr = (unsigned long long)(uintptr_t)(wsh + WS_ELR  + (size_t)j * 4096);
        const unsigned lWr = lds_off_of(sWr), lEl = lds_off_of(sEl), lEr = lds_off_of(sElr);
#pragma unroll
        for (int s = 0; s < 2; ++s) {                 // 4096 f16 = 2 sweeps x 256 thr x 16B
            const unsigned p = (unsigned)tid * 8u + (unsigned)s * 2048u;  // f16 idx
            async_b128(lWr + p * 2, bWr, p * 2);
            async_b128(lEl + p * 2, bEl, p * 2);
            async_b128(lEr + p * 2, bEr, p * 2);
        }
        if (j != 0) {
            const unsigned long long bWf =
                (unsigned long long)(uintptr_t)(wsh + WS_WFWD + (size_t)(j - 1) * 4096);
            const unsigned lWf = lds_off_of(sW0);
#pragma unroll
            for (int s = 0; s < 2; ++s) {
                const unsigned p = (unsigned)tid * 8u + (unsigned)s * 2048u;
                async_b128(lWf + p * 2, bWf, p * 2);
            }
        }
    } else {
        const float* wr  = W_rec_w + (size_t)j * 4096;
        const float* el  = E_l     + (size_t)j * 4096;
        const float* elr = E_l_r   + (size_t)j * 4096;
        for (int i4 = tid; i4 < 1024; i4 += NTHREADS) {
            v4f a = ((const v4f*)wr)[i4], b = ((const v4f*)el)[i4], c = ((const v4f*)elr)[i4];
            v4h ha, hb, hc;
#pragma unroll
            for (int k = 0; k < 4; ++k) { ha[k] = (_Float16)a[k]; hb[k] = (_Float16)b[k]; hc[k] = (_Float16)c[k]; }
            ((v4h*)sWr)[i4] = ha; ((v4h*)sEl)[i4] = hb; ((v4h*)sElr)[i4] = hc;
        }
        if (j != 0) {
            const float* wf = W_fwd_w + (size_t)(j - 1) * 4096;
            for (int i4 = tid; i4 < 1024; i4 += NTHREADS) {
                v4f a = ((const v4f*)wf)[i4];
                v4h h;
#pragma unroll
                for (int k = 0; k < 4; ++k) h[k] = (_Float16)a[k];
                ((v4h*)sW0)[i4] = h;
            }
        }
    }

    // ---------------- stage y tiles (sync f32 -> f16, float4) ---------------
    for (int i4 = tid; i4 < BT * KDIM / 4; i4 += NTHREADS) {
        const int idx = i4 * 4;                  // element index in tile
        const int r = idx >> 6, c = idx & 63;
        v4f a = *(const v4f*)(y + (size_t)(b0 + r) * NTOT + j * KDIM + c);
        v4h h;
#pragma unroll
        for (int k = 0; k < 4; ++k) h[k] = (_Float16)a[k];
        ((v4h*)sY)[i4] = h;
    }
    if (j != 0) {
        for (int i4 = tid; i4 < BT * KDIM / 4; i4 += NTHREADS) {
            const int idx = i4 * 4;
            const int r = idx >> 6, c = idx & 63;
            v4f a = *(const v4f*)(y + (size_t)(b0 + r) * NTOT + (j - 1) * KDIM + c);
            v4h h;
#pragma unroll
            for (int k = 0; k < 4; ++k) h[k] = (_Float16)a[k];
            ((v4h*)sA0)[i4] = h;
        }
    }
    if (use_ws) async_wait0();
    __syncthreads();

    // ---------------- phase A: net_out / net_rec ----------------------------
    v8f acc_o[4], acc_r[4];
#pragma unroll
    for (int nt = 0; nt < 4; ++nt) {
        const int n0 = nt * 16;
        const float b_out = (j == 0) ? W_in_b[n0 + nl]
                                     : W_fwd_b[(size_t)(j - 1) * KDIM + n0 + nl];
        const float b_rec = W_rec_b[(size_t)j * KDIM + n0 + nl];
#pragma unroll
        for (int v = 0; v < 8; ++v) { acc_o[nt][v] = b_out; acc_r[nt][v] = b_rec; }
        acc_r[nt] = wmma_acc(sY, KDIM, sWr, KDIM, m0, n0, KDIM, lane, acc_r[nt]);
    }

    if (j == 0) {
        // input GEMM, K = 256 streamed in 64-wide chunks through sW0/sA0
        for (int kc = 0; kc < IN_DIM_; kc += KDIM) {
            __syncthreads();                       // chunk buffers free
            if (use_ws) {
                const unsigned long long bW = (unsigned long long)(uintptr_t)(wsh + WS_WIN);
                const unsigned long long bU = (unsigned long long)(uintptr_t)(wsh + WS_UT);
                const unsigned lW = lds_off_of(sW0), lU = lds_off_of(sA0);
#pragma unroll
                for (int s = 0; s < 2; ++s) {      // sW0: 4096 f16
                    const unsigned p = (unsigned)tid * 8u + (unsigned)s * 2048u;
                    const unsigned o = p >> 6, c = p & 63u;
                    async_b128(lW + p * 2, bW, (o * 256u + (unsigned)kc + c) * 2);
                }
#pragma unroll
                for (int s = 0; s < 4; ++s) {      // sA0: 8192 f16
                    const unsigned p = (unsigned)tid * 8u + (unsigned)s * 2048u;
                    const unsigned r = p >> 6, c = p & 63u;
                    async_b128(lU + p * 2, bU,
                               (((unsigned)b0 + r) * 256u + (unsigned)kc + c) * 2);
                }
                async_wait0();
            } else {
                for (int i4 = tid; i4 < 1024; i4 += NTHREADS) {
                    const int idx = i4 * 4;
                    const int o = idx >> 6, c = idx & 63;
                    v4f a = *(const v4f*)(W_in_w + (size_t)o * IN_DIM_ + kc + c);
                    v4h h;
#pragma unroll
                    for (int k = 0; k < 4; ++k) h[k] = (_Float16)a[k];
                    ((v4h*)sW0)[i4] = h;
                }
                for (int i4 = tid; i4 < 2048; i4 += NTHREADS) {
                    const int idx = i4 * 4;
                    const int r = idx >> 6, c = idx & 63;
                    v4f a = *(const v4f*)(u_t + (size_t)(b0 + r) * IN_DIM_ + kc + c);
                    v4h h;
#pragma unroll
                    for (int k = 0; k < 4; ++k) h[k] = (_Float16)a[k];
                    ((v4h*)sA0)[i4] = h;
                }
            }
            __syncthreads();
#pragma unroll
            for (int nt = 0; nt < 4; ++nt)
                acc_o[nt] = wmma_acc(sA0, KDIM, sW0, KDIM, m0, nt * 16, KDIM,
                                     lane, acc_o[nt]);
        }
    } else {
#pragma unroll
        for (int nt = 0; nt < 4; ++nt)
            acc_o[nt] = wmma_acc(sA0, KDIM, sW0, KDIM, m0, nt * 16, KDIM,
                                 lane, acc_o[nt]);
    }

    float no_f[4][8], nr_f[4][8];
#pragma unroll
    for (int nt = 0; nt < 4; ++nt) {
        const int n0 = nt * 16;
#pragma unroll
        for (int v = 0; v < 8; ++v) {
            const float no = tanhf(acc_o[nt][v]);
            const float nr = tanhf(acc_r[nt][v]);
            no_f[nt][v] = no;  nr_f[nt][v] = nr;
            const int m = m0 + mh + v;
            sNo[m * KDIM + n0 + nl] = (_Float16)no;
            sNr[m * KDIM + n0 + nl] = (_Float16)nr;
        }
    }
    __syncthreads();

    // ---------------- phase B: drive GEMMs + elementwise update -------------
#pragma unroll
    for (int nt = 0; nt < 4; ++nt) {
        const int n0 = nt * 16;
        v8f acc = {};
        acc = wmma_acc(sNo, KDIM, sEl,  KDIM, m0, n0, KDIM, lane, acc);
        acc = wmma_acc(sNr, KDIM, sElr, KDIM, m0, n0, KDIM, lane, acc);

        const float rtau = 1.0f /
            (softplusf(tau_raw[(size_t)j * KDIM + n0 + nl]) + TAU_EPS_);
#pragma unroll
        for (int v = 0; v < 8; ++v) {
            const int m = m0 + mh + v;
            const size_t g = (size_t)(b0 + m) * NTOT + (size_t)j * KDIM + n0 + nl;
            const float yv    = y[g];                                   // L2 hit
            const float decay = rtau + fabsf(no_f[nt][v]) + fabsf(nr_f[nt][v]);
            out[g] = (yv + DT_ * acc[v]) / (1.0f + DT_ * decay);
        }
    }
}

extern "C" void kernel_launch(void* const* d_in, const int* in_sizes, int n_in,
                              void* d_out, int out_size, void* d_ws, size_t ws_size,
                              hipStream_t stream) {
    const float* y       = (const float*)d_in[0];
    const float* u_t     = (const float*)d_in[1];
    const float* tau_raw = (const float*)d_in[2];
    const float* W_in_w  = (const float*)d_in[3];
    const float* W_in_b  = (const float*)d_in[4];
    const float* W_fwd_w = (const float*)d_in[5];
    const float* W_fwd_b = (const float*)d_in[6];
    const float* W_rec_w = (const float*)d_in[7];
    const float* W_rec_b = (const float*)d_in[8];
    const float* E_l     = (const float*)d_in[9];
    const float* E_l_r   = (const float*)d_in[10];
    float* out = (float*)d_out;

    _Float16* wsh  = (_Float16*)d_ws;
    const int use_ws = (ws_size >= WS_BYTES) ? 1 : 0;

    if (use_ws) {
        auto cvt = [&](const float* s, _Float16* d, int n) {
            const int n4 = n / 4;
            cvt_kernel<<<(n4 + 255) / 256, 256, 0, stream>>>(s, d, n4);
        };
        cvt(W_in_w,  wsh + WS_WIN,  64 * 256);
        cvt(W_fwd_w, wsh + WS_WFWD, 63 * 64 * 64);
        cvt(W_rec_w, wsh + WS_WREC, 64 * 64 * 64);
        cvt(E_l,     wsh + WS_EL,   64 * 64 * 64);
        cvt(E_l_r,   wsh + WS_ELR,  64 * 64 * 64);
        cvt(u_t,     wsh + WS_UT,   BATCH_ * IN_DIM_);
    }

    dim3 grid(NBLK, BATCH_ / BT, 1);   // 64 x 32 workgroups
    dim3 block(NTHREADS, 1, 1);        // 8 wave32
    ltcn_kernel<<<grid, block, 0, stream>>>(y, u_t, tau_raw, W_in_w, W_in_b,
                                            W_fwd_w, W_fwd_b, W_rec_w, W_rec_b,
                                            E_l, E_l_r, wsh, use_ws, out);
}
